// Linear_Classifier_26199300506063
// MI455X (gfx1250) — matrix operand order repaired and checked by reading, not compiled, not run on hardware
//
#include <hip/hip_runtime.h>
#include <stdint.h>

// ---------------------------------------------------------------------------
// logits[n,c] = sum_{k,i,j} head[n,k,i]*tail[n,k,j]*W[c,k*4096+i*64+j] + b[c]
// Fused bf16-WMMA GEMM: A = W tile (16c x 32k), B = on-the-fly outer-product
// feature tile (32k x 16n) built with v_pk_mul_bf16 (scalar row-scale of tail).
// ---------------------------------------------------------------------------

typedef __attribute__((ext_vector_type(16))) __bf16 v16bf;
typedef __attribute__((ext_vector_type(8)))  float  v8f;

#define EMB    768
#define BS     64
#define NBLK   12
#define NCLS   97
#define NPAD   112          // 7 tiles of 16 classes
#define KFLAT  (EMB * BS)   // 49152 flattened reduction dim
#define ROWS_PER_BLOCK 32   // two 16-row n-tiles per block
#define WAVES_PER_BLOCK 7   // one wave per class tile
#define THREADS (WAVES_PER_BLOCK * 32)

// float -> bf16 via round-to-nearest-even (pure integer, always compiles)
__device__ __forceinline__ unsigned short f32_to_bf16(float f) {
  union { float f; unsigned u; } v; v.f = f;
  unsigned r = v.u + 0x7FFFu + ((v.u >> 16) & 1u);
  return (unsigned short)(r >> 16);
}

// packed bf16 multiply: 2 lanes of bf16 per dword, 8 bf16 per uint4
__device__ __forceinline__ uint4 pkmul4(uint4 a, unsigned s) {
  uint4 d;
  asm("v_pk_mul_bf16 %0, %1, %2" : "=v"(d.x) : "v"(a.x), "v"(s));
  asm("v_pk_mul_bf16 %0, %1, %2" : "=v"(d.y) : "v"(a.y), "v"(s));
  asm("v_pk_mul_bf16 %0, %1, %2" : "=v"(d.z) : "v"(a.z), "v"(s));
  asm("v_pk_mul_bf16 %0, %1, %2" : "=v"(d.w) : "v"(a.w), "v"(s));
  return d;
}

union Frag { v16bf v; uint4 q[2]; };

// --- prep: W fp32 [97][49152] -> bf16 [112][49152], rows >= 97 zeroed -------
__global__ void convert_w_bf16(const float* __restrict__ W,
                               unsigned short* __restrict__ Wb) {
  long idx = (long)blockIdx.x * blockDim.x + threadIdx.x;
  if (idx >= (long)NPAD * KFLAT) return;
  int c = (int)(idx / KFLAT);
  int k = (int)(idx % KFLAT);
  float val = (c < NCLS) ? W[(long)c * KFLAT + k] : 0.0f;
  Wb[idx] = f32_to_bf16(val);
}

// --- main fused bilinear-classifier kernel ---------------------------------
__global__ __launch_bounds__(THREADS)
void bilinear_wmma_kernel(const unsigned short* __restrict__ Wb,
                          const float* __restrict__ head,
                          const float* __restrict__ tail,
                          const float* __restrict__ bias,
                          float* __restrict__ out) {
  // tail rows for this block, converted to bf16 (32 rows x 768 = 48 KB)
  __shared__ unsigned short sT[ROWS_PER_BLOCK * EMB];

  const int nbase = blockIdx.x * ROWS_PER_BLOCK;
  for (int idx = threadIdx.x; idx < ROWS_PER_BLOCK * EMB; idx += THREADS) {
    int row = idx / EMB, col = idx % EMB;
    sT[idx] = f32_to_bf16(tail[(long)(nbase + row) * EMB + col]);
  }
  __syncthreads();

  const int wave = threadIdx.x >> 5;   // class tile 0..6
  const int lane = threadIdx.x & 31;
  const int l = lane & 15;             // A: row (class); B/C: column (sample)
  const int h = lane >> 4;             // lane half selects K sub-range

  // A operand: one row of bf16 W for this lane's class
  const unsigned short* __restrict__ arow = Wb + (long)(wave * 16 + l) * KFLAT;
  // B operand sources: tail rows in LDS, head rows (scalars) from global
  const unsigned short* __restrict__ b2r0 = sT + l * EMB;
  const unsigned short* __restrict__ b2r1 = sT + (16 + l) * EMB;
  const float* __restrict__ h1r0 = head + (long)(nbase + l) * EMB;
  const float* __restrict__ h1r1 = head + (long)(nbase + 16 + l) * EMB;

  v8f acc0 = {};
  v8f acc1 = {};

  for (int k = 0; k < NBLK; ++k) {
    for (int i = 0; i < BS; ++i) {
      const int col = k * BS + i;
      unsigned s0 = f32_to_bf16(h1r0[col]); s0 |= s0 << 16;  // packed scalar
      unsigned s1 = f32_to_bf16(h1r1[col]); s1 |= s1 << 16;
      const int kbase = k * (BS * BS) + i * BS;              // kappa0 (jh=0)
#pragma unroll
      for (int jh = 0; jh < 2; ++jh) {
        const int kap = kbase + jh * 32;
        // A fragment (16x32 bf16): lanes 0-15 K=0..15, 16-31 K=16..31
        Frag A;
        A.q[0] = *(const uint4*)(arow + kap + h * 16);
        A.q[1] = *(const uint4*)(arow + kap + h * 16 + 8);
        // B fragment (32x16 bf16): lane column n=l, K = h*16..h*16+15
        const int j0 = k * BS + jh * 32 + h * 16;
        uint4 t0a = *(const uint4*)(b2r0 + j0);
        uint4 t0b = *(const uint4*)(b2r0 + j0 + 8);
        uint4 t1a = *(const uint4*)(b2r1 + j0);
        uint4 t1b = *(const uint4*)(b2r1 + j0 + 8);
        Frag B0, B1;
        B0.q[0] = pkmul4(t0a, s0); B0.q[1] = pkmul4(t0b, s0);
        B1.q[0] = pkmul4(t1a, s1); B1.q[1] = pkmul4(t1b, s1);
        // D = A x B + C  (fp32 accumulate); A fragment reused for both n-tiles
        acc0 = __builtin_amdgcn_wmma_f32_16x16x32_bf16(
            false, A.v, false, B0.v, (short)0, acc0, false, false);
        acc1 = __builtin_amdgcn_wmma_f32_16x16x32_bf16(
            false, A.v, false, B1.v, (short)0, acc1, false, false);
      }
    }
  }

  // epilogue: C layout -> lane column n=l, row class = r + h*8; add bias
  const int cbase = wave * 16 + h * 8;
#pragma unroll
  for (int r = 0; r < 8; ++r) {
    const int c = cbase + r;
    if (c < NCLS) {
      const float bb = bias[c];
      out[(long)(nbase + l)      * NCLS + c] = acc0[r] + bb;
      out[(long)(nbase + 16 + l) * NCLS + c] = acc1[r] + bb;
    }
  }
}

// ---------------------------------------------------------------------------
extern "C" void kernel_launch(void* const* d_in, const int* in_sizes, int n_in,
                              void* d_out, int out_size, void* d_ws, size_t ws_size,
                              hipStream_t stream) {
  const float* head = (const float*)d_in[0];   // [N,768] fp32
  const float* tail = (const float*)d_in[1];   // [N,768] fp32
  const float* W    = (const float*)d_in[2];   // [97,49152] fp32
  const float* bias = (const float*)d_in[3];   // [97] fp32
  float* out = (float*)d_out;                  // [N,97] fp32

  const int n = in_sizes[0] / EMB;             // 4096

  // workspace: bf16 copy of W, zero-padded to 112 classes (11 MB, L2-resident)
  unsigned short* Wb = (unsigned short*)d_ws;

  const long totW = (long)NPAD * KFLAT;
  convert_w_bf16<<<(int)((totW + 255) / 256), 256, 0, stream>>>(W, Wb);

  bilinear_wmma_kernel<<<n / ROWS_PER_BLOCK, THREADS, 0, stream>>>(
      Wb, head, tail, bias, out);
}
